// HeteroGAT_14328010900294
// MI455X (gfx1250) — matrix operand order repaired and checked by reading, not compiled, not run on hardware
//
#include <hip/hip_runtime.h>
#include <hip/hip_bf16.h>

#define HID 128

typedef __attribute__((ext_vector_type(16))) __bf16 v16bf;
typedef __attribute__((ext_vector_type(8)))  float  v8f;

// ---------- helpers ----------
static __device__ __forceinline__ unsigned fenc(float f) {
  // monotonic float->uint mapping so atomicMax(u32) == float max (handles negatives)
  unsigned u = __float_as_uint(f);
  return (u & 0x80000000u) ? ~u : (u | 0x80000000u);
}
static __device__ __forceinline__ float fdec(unsigned u) {
  return __uint_as_float((u & 0x80000000u) ? (u & 0x7fffffffu) : ~u);
}
static __device__ __forceinline__ float leakyrelu(float a) { return a > 0.f ? a : 0.2f * a; }
static __device__ __forceinline__ float elu1(float v) { return v > 0.f ? v : (__expf(v) - 1.f); }

// ---------- weight pre-pack: f32 W[K][N] -> bf16 hi/lo in WMMA B-fragment layout ----------
// packed element idx = ((kt*(N/16) + nt)*32 + lane)*16 + i
// fragment mapping (contiguous k, same order as the A fragment): lane=(half,r): K = kt*32 + half*16 + i ; N = nt*16 + r
__global__ void pack_w(const float* __restrict__ W, __bf16* __restrict__ hi,
                       __bf16* __restrict__ lo, int K, int N) {
  int idx = blockIdx.x * blockDim.x + threadIdx.x;
  if (idx >= K * N) return;
  int i    = idx & 15;
  int lane = (idx >> 4) & 31;
  int t    = idx >> 9;
  int ntiles = N >> 4;
  int kt = t / ntiles, nt = t - kt * ntiles;
  int half = lane >> 4, r = lane & 15;
  int k = kt * 32 + half * 16 + i;
  int n = nt * 16 + r;
  float f = W[(size_t)k * N + n];
  __bf16 h = (__bf16)f;
  hi[idx] = h;
  lo[idx] = (__bf16)(f - (float)h);
}

// ---------- WMMA GEMM: C[M,N] = act(A[M,K] @ W[K,N] + bias) ----------
// A: fp32, converted to bf16 hi/lo per fragment; W: pre-packed bf16 hi/lo fragments.
// 2-term split (ah*bh + ah*bl + al*bh) -> ~fp32 accuracy at WMMA rate.
// block = 128 threads (4 waves); each wave owns up to TWO 16x16 column tiles (t0, t0+4),
// reusing the A fragment -> 6 WMMAs per k-step per wave. grid.y = M/16, grid.x = 1.
__global__ void gemm_bias_act(const float* __restrict__ A, const __bf16* __restrict__ Whi,
                              const __bf16* __restrict__ Wlo, const float* __restrict__ bias,
                              float* __restrict__ C, int M, int K, int N, int act) {
  const int wid  = threadIdx.x >> 5;
  const int lane = threadIdx.x & 31;
  const int half = lane >> 4;
  const int r    = lane & 15;
  const int ntiles = N >> 4;
  const int span = (int)(gridDim.x * (blockDim.x >> 5));
  const int t0 = (int)blockIdx.x * (blockDim.x >> 5) + wid;
  const int t1 = t0 + span;
  if (t0 >= ntiles) return;                         // wave-uniform; EXEC stays all-1s
  const bool has1 = (t1 < ntiles);
  const int m0 = blockIdx.y * 16;

  const float* __restrict__ arow = A + (size_t)(m0 + r) * K;
  v8f acc0 = {}, acc1 = {};
  const int ksteps = K >> 5;
  for (int kt = 0; kt < ksteps; ++kt) {
    const int k0 = kt * 32;
    __builtin_prefetch(arow + k0 + 32, 0, 3);       // global_prefetch_b8 (speculative OK)
    v16bf ah, al;
    // A fragment 16x32: e0..7 -> K = k0 + half*16 + i ; e8..15 -> K = k0 + half*16 + 8 + i
#pragma unroll
    for (int i = 0; i < 8; ++i) {
      float fa = arow[k0 + half * 16 + i];
      __bf16 h = (__bf16)fa;
      ah[i] = h; al[i] = (__bf16)(fa - (float)h);
    }
#pragma unroll
    for (int i = 0; i < 8; ++i) {
      float fa = arow[k0 + half * 16 + 8 + i];
      __bf16 h = (__bf16)fa;
      ah[8 + i] = h; al[8 + i] = (__bf16)(fa - (float)h);
    }
    // B fragments: contiguous 32B per lane (pre-packed), no conversions in the loop
    size_t o0 = ((((size_t)kt * ntiles + t0) << 5) + lane) << 4;
    v16bf bh0 = *(const v16bf*)(Whi + o0);
    v16bf bl0 = *(const v16bf*)(Wlo + o0);
    acc0 = __builtin_amdgcn_wmma_f32_16x16x32_bf16(false, ah, false, bh0, (short)0, acc0, false, false);
    acc0 = __builtin_amdgcn_wmma_f32_16x16x32_bf16(false, ah, false, bl0, (short)0, acc0, false, false);
    acc0 = __builtin_amdgcn_wmma_f32_16x16x32_bf16(false, al, false, bh0, (short)0, acc0, false, false);
    if (has1) {
      size_t o1 = ((((size_t)kt * ntiles + t1) << 5) + lane) << 4;
      v16bf bh1 = *(const v16bf*)(Whi + o1);
      v16bf bl1 = *(const v16bf*)(Wlo + o1);
      acc1 = __builtin_amdgcn_wmma_f32_16x16x32_bf16(false, ah, false, bh1, (short)0, acc1, false, false);
      acc1 = __builtin_amdgcn_wmma_f32_16x16x32_bf16(false, ah, false, bl1, (short)0, acc1, false, false);
      acc1 = __builtin_amdgcn_wmma_f32_16x16x32_bf16(false, al, false, bh1, (short)0, acc1, false, false);
    }
  }
  // epilogue (branch-free bias; C layout: VGPR v -> row m0 + half*8 + v, col = tile*16 + r)
  {
    const int n0 = t0 * 16;
    const float bv = bias[n0 + r];
#pragma unroll
    for (int v = 0; v < 8; ++v) {
      float val = acc0[v] + bv;
      if (act) val = elu1(val);
      C[(size_t)(m0 + half * 8 + v) * N + (n0 + r)] = val;
    }
  }
  if (has1) {
    const int n1 = t1 * 16;
    const float bv = bias[n1 + r];
#pragma unroll
    for (int v = 0; v < 8; ++v) {
      float val = acc1[v] + bv;
      if (act) val = elu1(val);
      C[(size_t)(m0 + half * 8 + v) * N + (n1 + r)] = val;
    }
  }
}

// ---------- per-node attention scores: asrc[n]=h[n,:]·a_s, adst[n]=h[n,:]·a_d ----------
__global__ void att_scores(const float* __restrict__ h, const float* __restrict__ a_s,
                           const float* __restrict__ a_d, float* __restrict__ asrc,
                           float* __restrict__ adst, int Nn) {
  int node = blockIdx.x * (blockDim.x >> 5) + (threadIdx.x >> 5);
  if (node >= Nn) return;
  int lane = threadIdx.x & 31;
  const float* row = h + (size_t)node * HID;
  float s1 = 0.f, s2 = 0.f;
  for (int i = lane; i < HID; i += 32) { float v = row[i]; s1 += v * a_s[i]; s2 += v * a_d[i]; }
  for (int off = 16; off > 0; off >>= 1) { s1 += __shfl_down(s1, off); s2 += __shfl_down(s2, off); }
  if (lane == 0) { asrc[node] = s1; adst[node] = s2; }
}

// ---------- edge softmax passes ----------
__global__ void edge_init(const float* __restrict__ asrc, const float* __restrict__ adst,
                          float* __restrict__ selfa, unsigned* __restrict__ menc, int Nn) {
  int n = blockIdx.x * blockDim.x + threadIdx.x;
  if (n >= Nn) return;
  float a = leakyrelu(asrc[n] + adst[n]);           // self-loop alpha
  selfa[n] = a;
  menc[n]  = fenc(a);                               // max init includes self loop
}

__global__ void edge_max(const int* __restrict__ src, const int* __restrict__ dst,
                         const float* __restrict__ asrc, const float* __restrict__ adst,
                         unsigned* __restrict__ menc, int E) {
  int e = blockIdx.x * blockDim.x + threadIdx.x;
  if (e >= E) return;
  float a = leakyrelu(asrc[src[e]] + adst[dst[e]]);
  atomicMax(&menc[dst[e]], fenc(a));                // GLOBAL_ATOMIC_MAX_U32, L2-resident
}

__global__ void ssum_init(const float* __restrict__ selfa, const unsigned* __restrict__ menc,
                          float* __restrict__ ssum, int Nn) {
  int n = blockIdx.x * blockDim.x + threadIdx.x;
  if (n >= Nn) return;
  ssum[n] = __expf(selfa[n] - fdec(menc[n]));
}

__global__ void edge_sum(const int* __restrict__ src, const int* __restrict__ dst,
                         const float* __restrict__ asrc, const float* __restrict__ adst,
                         const unsigned* __restrict__ menc, float* __restrict__ ssum, int E) {
  int e = blockIdx.x * blockDim.x + threadIdx.x;
  if (e >= E) return;
  int d = dst[e];
  float a = leakyrelu(asrc[src[e]] + adst[d]);
  atomicAdd(&ssum[d], __expf(a - fdec(menc[d])));
}

// out[n,:] = h[n,:] * w_self[n] + bias   (self-loop term initializes the segment sum)
__global__ void msg_init(const float* __restrict__ h, const float* __restrict__ selfa,
                         const unsigned* __restrict__ menc, const float* __restrict__ ssum,
                         const float* __restrict__ bias, float* __restrict__ msg, int Nn) {
  int idx = blockIdx.x * blockDim.x + threadIdx.x;
  if (idx >= Nn * HID) return;
  int n = idx >> 7, c = idx & (HID - 1);
  float w = __expf(selfa[n] - fdec(menc[n])) / (ssum[n] + 1e-16f);
  msg[idx] = h[idx] * w + bias[c];
}

// one wave per edge: 32 lanes x float4 = 128-col row; atomic f32 adds land in L2
__global__ void edge_scatter(const int* __restrict__ src, const int* __restrict__ dst,
                             const float* __restrict__ asrc, const float* __restrict__ adst,
                             const unsigned* __restrict__ menc, const float* __restrict__ ssum,
                             const float* __restrict__ h, float* __restrict__ out, int E) {
  int e = blockIdx.x * (blockDim.x >> 5) + (threadIdx.x >> 5);
  if (e >= E) return;
  int lane = threadIdx.x & 31;
  int s = src[e], d = dst[e];
  float a = leakyrelu(asrc[s] + adst[d]);
  float w = __expf(a - fdec(menc[d])) / (ssum[d] + 1e-16f);
  const float4 hv = *(const float4*)(h + (size_t)s * HID + lane * 4);
  float* orow = out + (size_t)d * HID + lane * 4;
  atomicAdd(orow + 0, hv.x * w);
  atomicAdd(orow + 1, hv.y * w);
  atomicAdd(orow + 2, hv.z * w);
  atomicAdd(orow + 3, hv.w * w);
}

// ---------- semantic attention (user: T=2) + optional residual ----------
__global__ void combine_user(const float* __restrict__ m0, const float* __restrict__ m1,
                             const float* __restrict__ taW, const float* __restrict__ tab,
                             const float* __restrict__ xprev, int use_res,
                             float* __restrict__ agg, int Nn) {
  int node = blockIdx.x * (blockDim.x >> 5) + (threadIdx.x >> 5);
  if (node >= Nn) return;
  int lane = threadIdx.x & 31;
  const float* r0 = m0 + (size_t)node * HID;
  const float* r1 = m1 + (size_t)node * HID;
  float s0 = 0.f, s1 = 0.f;
  for (int i = lane; i < HID; i += 32) { float w = taW[i]; s0 += r0[i] * w; s1 += r1[i] * w; }
  for (int off = 16; off > 0; off >>= 1) { s0 += __shfl_down(s0, off); s1 += __shfl_down(s1, off); }
  s0 = __shfl(s0, 0) + tab[0];
  s1 = __shfl(s1, 0) + tab[0];
  float mx = fmaxf(s0, s1);
  float e0 = __expf(s0 - mx), e1 = __expf(s1 - mx);
  float inv = 1.f / (e0 + e1);
  float w0 = e0 * inv, w1 = e1 * inv;
  float* ao = agg + (size_t)node * HID;
  const float* xp = xprev + (size_t)node * HID;
  for (int i = lane; i < HID; i += 32) {
    float v = w0 * r0[i] + w1 * r1[i];
    if (use_res) v += xp[i];
    ao[i] = v;
  }
}

__global__ void residual_add(float* __restrict__ x, const float* __restrict__ r, int n) {
  int i = blockIdx.x * blockDim.x + threadIdx.x;
  if (i < n) x[i] += r[i];
}

// ---------- batch norm ----------
__global__ void zero2(float* a, float* b) { if (threadIdx.x < HID) { a[threadIdx.x] = 0.f; b[threadIdx.x] = 0.f; } }

#define BN_ROWS 250
__global__ void bn_stats(const float* __restrict__ x, float* __restrict__ sums,
                         float* __restrict__ sqs, int Nn) {
  int c = threadIdx.x;                               // 128 threads = 128 channels
  int r0 = blockIdx.x * BN_ROWS;
  int r1 = r0 + BN_ROWS; if (r1 > Nn) r1 = Nn;
  float s = 0.f, q = 0.f;
  for (int r = r0; r < r1; ++r) { float v = x[(size_t)r * HID + c]; s += v; q += v * v; }
  atomicAdd(&sums[c], s);
  atomicAdd(&sqs[c], q);
}

__global__ void bn_apply(const float* __restrict__ x, const float* __restrict__ sums,
                         const float* __restrict__ sqs, const float* __restrict__ g,
                         const float* __restrict__ b, float* __restrict__ out, int Nn) {
  int idx = blockIdx.x * blockDim.x + threadIdx.x;
  if (idx >= Nn * HID) return;
  int c = idx & (HID - 1);
  float inv = 1.f / (float)Nn;
  float mean = sums[c] * inv;
  float var  = sqs[c] * inv - mean * mean;
  float v = g[c] * (x[idx] - mean) * rsqrtf(var + 1e-5f) + b[c];
  out[idx] = elu1(v);
}

// ---------- host orchestration ----------
extern "C" void kernel_launch(void* const* d_in, const int* in_sizes, int n_in,
                              void* d_out, int out_size, void* d_ws, size_t ws_size,
                              hipStream_t stream) {
  const int Nn = in_sizes[0] / 64;                   // 20000
  const int E  = in_sizes[2] / 2;                    // 320000
  const float* x_user = (const float*)d_in[0];
  const float* x_item = (const float*)d_in[1];
  const int* esrc[3] = {(const int*)d_in[2], (const int*)d_in[3], (const int*)d_in[4]};
  const int* edst[3] = {esrc[0] + E, esrc[1] + E, esrc[2] + E};
  auto P = [&](int i) { return (const float*)d_in[i]; };

  // workspace carve (entire model state ~90MB -> fits the 192MB L2)
  char* wp = (char*)d_ws;
  auto carveB = [&](size_t bytes) { void* p = wp; wp += ((bytes + 255) & ~(size_t)255); return p; };
  auto carve  = [&](size_t nf) { return (float*)carveB(nf * 4); };
  const size_t NH = (size_t)Nn * HID;
  float* xu[2] = {carve(NH), carve(NH)};
  float* xi[2] = {carve(NH), carve(NH)};
  float* h     = carve(NH);
  float* msgI0 = carve(NH);
  float* msgU0 = carve(NH);
  float* msgU1 = carve(NH);
  float* clfh  = carve((size_t)Nn * 64);
  float* asrc = carve(Nn), *adst = carve(Nn), *selfa = carve(Nn), *ssum = carve(Nn);
  unsigned* menc = (unsigned*)carve(Nn);
  float* bnsum = carve(HID), *bnsq = carve(HID);
  float* zbias = carve(HID);

  // packed weights: [0]=enc_user(64x128) [1]=enc_item(128x128) [2..10]=conv W (128x128)
  // [11]=clf_W1(128x64) [12]=clf_W2(64x16)
  __bf16* Whi[13]; __bf16* Wlo[13];
  int wK[13], wN[13]; const float* wSrc[13];
  wK[0] = 64;  wN[0] = HID; wSrc[0] = P(5);
  wK[1] = HID; wN[1] = HID; wSrc[1] = P(7);
  for (int l = 0; l < 3; ++l)
    for (int j = 0; j < 3; ++j) { int m = 2 + l * 3 + j; wK[m] = HID; wN[m] = HID; wSrc[m] = P(9 + 16 * l + 4 * j); }
  wK[11] = HID; wN[11] = 64; wSrc[11] = P(61);
  wK[12] = 64;  wN[12] = 16; wSrc[12] = P(63);
  for (int m = 0; m < 13; ++m) {
    size_t kn = (size_t)wK[m] * wN[m];
    Whi[m] = (__bf16*)carveB(kn * 2);
    Wlo[m] = (__bf16*)carveB(kn * 2);
    pack_w<<<(unsigned)((kn + 255) / 256), 256, 0, stream>>>(wSrc[m], Whi[m], Wlo[m], wK[m], wN[m]);
  }
  zero2<<<1, 128, 0, stream>>>(zbias, zbias);        // zero bias vector for the h-projection GEMMs

  auto gemm = [&](const float* A, int m, const float* bias, float* C, int M, int act) {
    dim3 grid(1u, (unsigned)(M / 16));               // 4 waves/block, up to 2 column tiles per wave
    gemm_bias_act<<<grid, dim3(128), 0, stream>>>(A, Whi[m], Wlo[m], bias, C, M, wK[m], wN[m], act);
  };
  const unsigned gN  = (unsigned)((Nn + 255) / 256);
  const unsigned gNw = (unsigned)((Nn + 7) / 8);
  const unsigned gE  = (unsigned)((E + 255) / 256);
  const unsigned gEw = (unsigned)((E + 7) / 8);
  const unsigned gNH = (unsigned)((NH + 255) / 256);

  // encoder: elu(x @ W + b)
  gemm(x_user, 0, P(6), xu[0], Nn, 1);
  gemm(x_item, 1, P(8), xi[0], Nn, 1);

  int cur = 0;
  for (int l = 0; l < 3; ++l) {
    const int base = 9 + 16 * l;
    const int   srcIsItem[3] = {0, 1, 0};            // buys, bought_by, follows
    float*      msgOf[3]     = {msgI0, msgU0, msgU1};
    for (int j = 0; j < 3; ++j) {
      const float* as = P(base + 4 * j + 1);
      const float* ad = P(base + 4 * j + 2);
      const float* bc = P(base + 4 * j + 3);
      const float* xs = srcIsItem[j] ? xi[cur] : xu[cur];
      gemm(xs, 2 + l * 3 + j, zbias, h, Nn, 0);
      att_scores  <<<gNw, 256, 0, stream>>>(h, as, ad, asrc, adst, Nn);
      edge_init   <<<gN,  256, 0, stream>>>(asrc, adst, selfa, menc, Nn);
      edge_max    <<<gE,  256, 0, stream>>>(esrc[j], edst[j], asrc, adst, menc, E);
      ssum_init   <<<gN,  256, 0, stream>>>(selfa, menc, ssum, Nn);
      edge_sum    <<<gE,  256, 0, stream>>>(esrc[j], edst[j], asrc, adst, menc, ssum, E);
      msg_init    <<<gNH, 256, 0, stream>>>(h, selfa, menc, ssum, bc, msgOf[j], Nn);
      edge_scatter<<<gEw, 256, 0, stream>>>(esrc[j], edst[j], asrc, adst, menc, ssum, h, msgOf[j], E);
    }
    // user: semantic attention over {bought_by, follows} (+ residual), into h
    combine_user<<<gNw, 256, 0, stream>>>(msgU0, msgU1, P(57), P(58), xu[cur], l > 0 ? 1 : 0, h, Nn);
    // item: T=1 -> softmax weight == 1 -> agg = msgI0 (+ residual in place)
    if (l > 0) residual_add<<<gNH, 256, 0, stream>>>(msgI0, xi[cur], (int)NH);
    // BN + ELU (batch statistics) per type
    zero2   <<<1, 128, 0, stream>>>(bnsum, bnsq);
    bn_stats<<<(unsigned)((Nn + BN_ROWS - 1) / BN_ROWS), 128, 0, stream>>>(h, bnsum, bnsq, Nn);
    bn_apply<<<gNH, 256, 0, stream>>>(h, bnsum, bnsq, P(base + 12), P(base + 13), xu[1 - cur], Nn);
    zero2   <<<1, 128, 0, stream>>>(bnsum, bnsq);
    bn_stats<<<(unsigned)((Nn + BN_ROWS - 1) / BN_ROWS), 128, 0, stream>>>(msgI0, bnsum, bnsq, Nn);
    bn_apply<<<gNH, 256, 0, stream>>>(msgI0, bnsum, bnsq, P(base + 14), P(base + 15), xi[1 - cur], Nn);
    cur ^= 1;
  }

  // classifier on user nodes
  gemm(xu[cur], 11, P(62), clfh, Nn, 1);
  gemm(clfh,    12, P(64), (float*)d_out, Nn, 0);
}
